// ConvAttention_2078764171350
// MI455X (gfx1250) — matrix operand order repaired and checked by reading, not compiled, not run on hardware
//
#include <hip/hip_runtime.h>
#include <hip/hip_bf16.h>

// ---------------------------------------------------------------------------
// CDNA5 (gfx1250) wave32 WMMA implementation of ConvAttention.
// All conv1d's + QK run on v_wmma_f32_16x16x32_f16 (f16 in, f32 accumulate).
// Round 4: padded-row strides are template constants -> the transposed B
// loads become one base address + 16 immediate-offset clause loads
// (no v_add_nc_u64 chains, no s_wait_xcnt serialization).
// ---------------------------------------------------------------------------

typedef __attribute__((ext_vector_type(16))) _Float16 v16h;
typedef __attribute__((ext_vector_type(8)))  _Float16 v8h;
typedef __attribute__((ext_vector_type(8)))  float    v8f;

union H16 { v16h v; v8h h[2]; };

#define BM 64
#define BN 64
#define BK 32
#define LDK 40          // padded LDS row (halves): 80 bytes, 16B aligned
#define NTHREADS 128    // 4 wave32 per block
#define COL0 8          // padded tensors: logical t=0 lives at column 8
#define NP_K 272        // padded row stride for Tk=200 tensors (256+16)
#define NP_Q 848        // padded row stride for Tq=800 tensors (832+16)
#define RAWM 832
#define RAWN 256

// ---- weights f16 [M][K] (K padded, zero-filled) -> LDS [m][k]; branch-free.
// Rows >= M are clamped (their C rows are forced to zero in the epilogue).
__device__ inline void load_a_pad(const _Float16* __restrict__ Atap,
                                  _Float16* Ash,
                                  int m0, int k0, int M, int K, int tid) {
  int row = tid & 63;
  int kh  = (tid >> 6) << 4;          // 0 or 16
  int gm  = m0 + row;
  if (gm >= M) gm = M - 1;            // clamp: safe, C row forced to zero
  const _Float16* p = Atap + gm * K + (k0 + kh);
  v8h v0 = *(const v8h*)p;
  v8h v1 = *(const v8h*)(p + 8);
  _Float16* dst = Ash + row * LDK + kh;
  *(v8h*)dst       = v0;
  *(v8h*)(dst + 8) = v1;
}

// ---- padded f16 [Krows][NP] -> LDS transposed [n][k]; branch-free.
// NP is a compile-time constant so the 16 strided loads fold into one base
// address + immediate offsets (single clause, no address-chain waits).
template <int NP>
__device__ inline void load_bt_pad(const _Float16* __restrict__ Bb,
                                   _Float16* Bsh,
                                   int n0, int k0, int shift, int tid) {
  int n  = tid & 63;
  int kh = (tid >> 6) << 4;           // 0 or 16
  const _Float16* p = Bb + (k0 + kh) * NP + (COL0 + n0 + n + shift);
  v8h v0, v1;
#pragma unroll
  for (int i = 0; i < 8; ++i) v0[i] = p[i * NP];
#pragma unroll
  for (int i = 0; i < 8; ++i) v1[i] = p[(i + 8) * NP];
  _Float16* dst = Bsh + n * LDK + kh;
  *(v8h*)dst       = v0;
  *(v8h*)(dst + 8) = v1;
}

// ---- one 64x64x32 block-tile step: 2x2 WMMA per wave -----------------------
__device__ inline void mma_step(const _Float16* Ash, const _Float16* Bsh,
                                int wm, int wn, int lane, v8f acc[2][2]) {
  int lm = lane & 15;
  int ka = (lane < 16) ? 0 : 8;    // A 16x32 f16 lane layout (ISA 7.12.2)
  int kb = (lane < 16) ? 0 : 8;    // B 32x16 f16 lane layout
  H16 a[2], b[2];
#pragma unroll
  for (int i = 0; i < 2; ++i) {
    const _Float16* p = Ash + (wm + i * 16 + lm) * LDK + ka;
    a[i].h[0] = *(const v8h*)(p);        // K = ka+0..7
    a[i].h[1] = *(const v8h*)(p + 16);   // K = ka+16..23
  }
#pragma unroll
  for (int j = 0; j < 2; ++j) {
    const _Float16* p = Bsh + (wn + j * 16 + lm) * LDK + kb;
    b[j].h[0] = *(const v8h*)(p);        // K = kb+0..7
    b[j].h[1] = *(const v8h*)(p + 16);   // K = kb+16..23
  }
#pragma unroll
  for (int i = 0; i < 2; ++i)
#pragma unroll
    for (int j = 0; j < 2; ++j)
      acc[i][j] = __builtin_amdgcn_wmma_f32_16x16x32_f16(
          false, a[i].v, false, b[j].v, (short)0, acc[i][j], false, false);
}

// ---- batched conv-as-GEMM on padded tensors:
// C[b,m,t] = bias[m] + sum_{tap,k} A[tap][m][k] * B[b][k][t+tap-taps/2]
// Epilogue writes zeros outside (M x Nvalid) -> output is itself padded. ----
template <bool RELU, int NP>
__global__ __launch_bounds__(NTHREADS)
void wmma_conv_gemm(const _Float16* __restrict__ A, const _Float16* __restrict__ Bp,
                    const float* __restrict__ bias, _Float16* __restrict__ Cp,
                    int M, int K, int Nvalid,
                    int rowsB, int rowsC, int taps) {
  __shared__ alignas(16) _Float16 Ash[2][BM * LDK];
  __shared__ alignas(16) _Float16 Bsh[2][BN * LDK];
  int tid = threadIdx.x;
  int lane = tid & 31, wave = tid >> 5;
  int wm = (wave >> 1) * 32, wn = (wave & 1) * 32;
  int n0 = blockIdx.x * BN;
  int m0 = blockIdx.y * BM;
  int bb = blockIdx.z;
  const _Float16* Bb = Bp + (size_t)bb * rowsB * NP;
  _Float16* Cb = Cp + (size_t)bb * rowsC * NP;
  int half_taps = taps >> 1;

  v8f acc[2][2];
#pragma unroll
  for (int i = 0; i < 2; ++i)
#pragma unroll
    for (int j = 0; j < 2; ++j)
#pragma unroll
      for (int r = 0; r < 8; ++r) acc[i][j][r] = 0.f;

  auto do_load = [&](int tap, int k0, int buf) {
    load_a_pad(A + tap * M * K, Ash[buf], m0, k0, M, K, tid);
    load_bt_pad<NP>(Bb, Bsh[buf], n0, k0, tap - half_taps, tid);
  };

  // double-buffered pipeline: load step s+1 before MMA on step s
  do_load(0, 0, 0);
  int tap = 0, k0 = 0, buf = 0;
  while (true) {
    __syncthreads();
    int tapn = tap, k0n = k0 + BK;
    if (k0n >= K) { k0n = 0; ++tapn; }
    bool more = (tapn < taps);
    if (more) do_load(tapn, k0n, buf ^ 1);
    mma_step(Ash[buf], Bsh[buf], wm, wn, lane, acc);
    if (!more) break;
    tap = tapn; k0 = k0n; buf ^= 1;
  }

  int lm = lane & 15;
  int rbase = (lane < 16) ? 0 : 8;   // C 16x16 f32 lane layout (ISA 7.12.2)
#pragma unroll
  for (int i = 0; i < 2; ++i)
#pragma unroll
    for (int j = 0; j < 2; ++j) {
      int col = n0 + wn + j * 16 + lm;
#pragma unroll
      for (int r = 0; r < 8; ++r) {
        int row = m0 + wm + i * 16 + rbase + r;
        float bv = bias[row < M ? row : 0];
        float v = acc[i][j][r] + bv;
        if (RELU) v = fmaxf(v, 0.f);
        if (row >= M || col >= Nvalid) v = 0.f;   // write the padding zeros
        Cb[row * NP + COL0 + col] = (_Float16)v;  // unconditional store
      }
    }
}

// ---- QK distance on padded tensors:
// rawP[b,q,t] = -5e-4*(q2[q] + k2[t] - 2*sum_c Q[c,q]K[c,t]); 832x256/batch.
template <int NPQ, int NPK>
__global__ __launch_bounds__(NTHREADS)
void wmma_qk_dist(const _Float16* __restrict__ Qp, const _Float16* __restrict__ Kp,
                  const float* __restrict__ q2, const float* __restrict__ k2,
                  float* __restrict__ rawP,
                  int K, int rowsQ, int rowsK) {
  __shared__ alignas(16) _Float16 Ash[2][BM * LDK];
  __shared__ alignas(16) _Float16 Bsh[2][BN * LDK];
  int tid = threadIdx.x;
  int lane = tid & 31, wave = tid >> 5;
  int wm = (wave >> 1) * 32, wn = (wave & 1) * 32;
  int n0 = blockIdx.x * BN;
  int m0 = blockIdx.y * BM;
  int bb = blockIdx.z;
  const _Float16* Qb = Qp + (size_t)bb * rowsQ * NPQ;
  const _Float16* Kb = Kp + (size_t)bb * rowsK * NPK;
  const float* q2b = q2 + (size_t)bb * RAWM;
  const float* k2b = k2 + (size_t)bb * RAWN;
  float* rawb = rawP + (size_t)bb * RAWM * RAWN;

  v8f acc[2][2];
#pragma unroll
  for (int i = 0; i < 2; ++i)
#pragma unroll
    for (int j = 0; j < 2; ++j)
#pragma unroll
      for (int r = 0; r < 8; ++r) acc[i][j][r] = 0.f;

  auto do_load = [&](int k0, int buf) {
    load_bt_pad<NPQ>(Qb, Ash[buf], m0, k0, 0, tid);  // A = Q^T tile
    load_bt_pad<NPK>(Kb, Bsh[buf], n0, k0, 0, tid);  // B = K tile
  };

  do_load(0, 0);
  int k0 = 0, buf = 0;
  while (true) {
    __syncthreads();
    int k0n = k0 + BK;
    bool more = (k0n < K);
    if (more) do_load(k0n, buf ^ 1);
    mma_step(Ash[buf], Bsh[buf], wm, wn, lane, acc);
    if (!more) break;
    k0 = k0n; buf ^= 1;
  }

  int lm = lane & 15;
  int rbase = (lane < 16) ? 0 : 8;
#pragma unroll
  for (int i = 0; i < 2; ++i)
#pragma unroll
    for (int j = 0; j < 2; ++j) {
      int col = n0 + wn + j * 16 + lm;
#pragma unroll
      for (int r = 0; r < 8; ++r) {
        int row = m0 + wm + i * 16 + rbase + r;
        float qk = acc[i][j][r];
        rawb[row * RAWN + col] = -5e-4f * (q2b[row] + k2b[col] - 2.f * qk);
      }
    }
}

// ---- per-(b,t) channel sum of squares over padded f16 tensor ---------------
__global__ void sumsq_pad(const _Float16* __restrict__ x, float* __restrict__ out,
                          int C, int T, int NP, int rowsP, int TP) {
  int t = blockIdx.x * blockDim.x + threadIdx.x;
  int b = blockIdx.y;
  if (t >= T) return;
  const _Float16* xb = x + (size_t)b * rowsP * NP + COL0;
  float s = 0.f;
  for (int c = 0; c < C; ++c) {
    float v = (float)xb[c * NP + t];
    s += v * v;
  }
  out[(size_t)b * TP + t] = s;
}

// ---- f32 (B,C,T) -> padded f16 [b][C..][NP] with column offset COL0 --------
__global__ void pad_convert(const float* __restrict__ src, _Float16* __restrict__ dst,
                            int C, int T, int NP, int rowsP) {
  int t = blockIdx.x * blockDim.x + threadIdx.x;
  int c = blockIdx.y;
  int b = blockIdx.z;
  if (t >= T) return;
  dst[((size_t)b * rowsP + c) * NP + COL0 + t] = (_Float16)src[((size_t)b * C + c) * T + t];
}

// ---- (Cout,Cin,taps) f32 -> [tap][Cout][KP] f16, zero-filled for ci>=Cin ---
__global__ void convert_weights(const float* __restrict__ w,
                                _Float16* __restrict__ out,
                                int Cout, int Cin, int taps, int KP) {
  int i = blockIdx.x * blockDim.x + threadIdx.x;
  int n = taps * Cout * KP;
  if (i >= n) return;
  int tap = i / (Cout * KP);
  int r   = i % (Cout * KP);
  int co  = r / KP;
  int ci  = r % KP;
  float v = (ci < Cin) ? w[(co * Cin + ci) * taps + tap] : 0.f;
  out[i] = (_Float16)v;
}

// ---- fused double softmax row kernel ---------------------------------------
// logprob = raw - logsumexp(raw) + log(prior + 1e-8)
// attn    = softmax( mask ? -inf : logprob )
__global__ __launch_bounds__(256)
void softmax_rows(const float* __restrict__ rawP, const float* __restrict__ prior,
                  const unsigned char* __restrict__ mask,
                  float* __restrict__ attn, float* __restrict__ logprob,
                  int Tq, int Tk) {
  __shared__ float red[256];
  const float NEG = -3.4e38f;
  int tid = threadIdx.x;
  int q = blockIdx.x, b = blockIdx.y;
  const float* rowp = rawP + ((size_t)b * RAWM + q) * RAWN;
  size_t rowoff = ((size_t)b * Tq + q) * Tk;
  bool valid = tid < Tk;

  float x = valid ? rowp[tid] : NEG;
  red[tid] = x; __syncthreads();
  for (int s = 128; s > 0; s >>= 1) {
    if (tid < s) red[tid] = fmaxf(red[tid], red[tid + s]);
    __syncthreads();
  }
  float mx = red[0]; __syncthreads();

  float e = valid ? __expf(x - mx) : 0.f;
  red[tid] = e; __syncthreads();
  for (int s = 128; s > 0; s >>= 1) {
    if (tid < s) red[tid] += red[tid + s];
    __syncthreads();
  }
  float lse = mx + __logf(red[0]); __syncthreads();

  float lp = 0.f, ml = NEG;
  if (valid) {
    lp = x - lse + __logf(prior[rowoff + tid] + 1e-8f);
    logprob[rowoff + tid] = lp;
    ml = mask[b * Tk + tid] ? NEG : lp;
  }

  red[tid] = ml; __syncthreads();
  for (int s = 128; s > 0; s >>= 1) {
    if (tid < s) red[tid] = fmaxf(red[tid], red[tid + s]);
    __syncthreads();
  }
  float mx2 = red[0]; __syncthreads();

  float e2 = valid ? __expf(ml - mx2) : 0.f;
  red[tid] = e2; __syncthreads();
  for (int s = 128; s > 0; s >>= 1) {
    if (tid < s) red[tid] += red[tid + s];
    __syncthreads();
  }
  float inv = 1.f / red[0];
  if (valid) attn[rowoff + tid] = e2 * inv;
}

// ---------------------------------------------------------------------------
extern "C" void kernel_launch(void* const* d_in, const int* in_sizes, int n_in,
                              void* d_out, int out_size, void* d_ws, size_t ws_size,
                              hipStream_t stream) {
  (void)in_sizes; (void)n_in; (void)out_size; (void)ws_size;

  const float* queries = (const float*)d_in[0];   // (16, 80, 800)
  const float* keys    = (const float*)d_in[1];   // (16, 512, 200)
  /* d_in[2] query_lens: unused by reference */
  const unsigned char* mask = (const unsigned char*)d_in[3];  // (16, 200, 1) bool
  const float* prior   = (const float*)d_in[4];   // (16, 800, 200)
  const float* kW1 = (const float*)d_in[5];
  const float* kb1 = (const float*)d_in[6];
  const float* kW2 = (const float*)d_in[7];
  const float* kb2 = (const float*)d_in[8];
  const float* qW1 = (const float*)d_in[9];
  const float* qb1 = (const float*)d_in[10];
  const float* qW2 = (const float*)d_in[11];
  const float* qb2 = (const float*)d_in[12];
  const float* qW3 = (const float*)d_in[13];
  const float* qb3 = (const float*)d_in[14];

  const int B = 16, Tq = 800, Tk = 200;

  // workspace carve-up (256B aligned); zero-region first (memset each call)
  char* ws = (char*)d_ws;
  size_t off = 0;
  auto carve = [&](size_t bytes) -> char* {
    char* p = ws + off;
    off = (off + bytes + 255) & ~(size_t)255;
    return p;
  };
  // ---- zeroed region: padded operand tensors + norm vectors ----
  _Float16* keysP    = (_Float16*)carve((size_t)B * 512  * NP_K * 2);
  _Float16* queriesP = (_Float16*)carve((size_t)B * 96   * NP_Q * 2);
  _Float16* kmidP    = (_Float16*)carve((size_t)B * 1024 * NP_K * 2);
  _Float16* q1P      = (_Float16*)carve((size_t)B * 192  * NP_Q * 2);
  _Float16* q2bP     = (_Float16*)carve((size_t)B * 128  * NP_Q * 2);
  _Float16* qencP    = (_Float16*)carve((size_t)B * 128  * NP_Q * 2);
  _Float16* kencP    = (_Float16*)carve((size_t)B * 128  * NP_K * 2);
  float*    q2s      = (float*)carve((size_t)B * RAWM * 4);
  float*    k2s      = (float*)carve((size_t)B * RAWN * 4);
  size_t zero_bytes = off;
  // ---- fully-overwritten region ----
  float*    rawP  = (float*)carve((size_t)B * RAWM * RAWN * 4);
  _Float16* kW1h  = (_Float16*)carve((size_t)3 * 1024 * 512 * 2);
  _Float16* kW2h  = (_Float16*)carve((size_t)80 * 1024 * 2);
  _Float16* qW1h  = (_Float16*)carve((size_t)3 * 160 * 96 * 2);
  _Float16* qW2h  = (_Float16*)carve((size_t)80 * 160 * 2);
  _Float16* qW3h  = (_Float16*)carve((size_t)80 * 96 * 2);

  float* attn_out = (float*)d_out;
  float* logp_out = attn_out + (size_t)B * Tq * Tk;

  // 0) zero the padded region (provides conv SAME zeros + K/N tails)
  hipMemsetAsync(ws, 0, zero_bytes, stream);

  // 1) pack inputs into padded f16
  pad_convert<<<dim3(1, 512, B), 256, 0, stream>>>(keys, keysP, 512, Tk, NP_K, 512);
  pad_convert<<<dim3(4, 80, B), 256, 0, stream>>>(queries, queriesP, 80, Tq, NP_Q, 96);

  // 2) weights -> f16 [tap][Cout][KP] (zero-filled K tails)
  {
    int n;
    n = 3 * 1024 * 512;
    convert_weights<<<(n + 255) / 256, 256, 0, stream>>>(kW1, kW1h, 1024, 512, 3, 512);
    n = 80 * 1024;
    convert_weights<<<(n + 255) / 256, 256, 0, stream>>>(kW2, kW2h, 80, 1024, 1, 1024);
    n = 3 * 160 * 96;
    convert_weights<<<(n + 255) / 256, 256, 0, stream>>>(qW1, qW1h, 160, 80, 3, 96);
    n = 80 * 160;
    convert_weights<<<(n + 255) / 256, 256, 0, stream>>>(qW2, qW2h, 80, 160, 1, 160);
    n = 80 * 96;
    convert_weights<<<(n + 255) / 256, 256, 0, stream>>>(qW3, qW3h, 80, 80, 1, 96);
  }

  // 3) keys path: relu(conv3(keys)) -> conv1 -> keys_enc
  wmma_conv_gemm<true, NP_K><<<dim3(4, 16, B), NTHREADS, 0, stream>>>(
      kW1h, keysP, kb1, kmidP, 1024, 512, Tk, 512, 1024, 3);
  wmma_conv_gemm<false, NP_K><<<dim3(4, 2, B), NTHREADS, 0, stream>>>(
      kW2h, kmidP, kb2, kencP, 80, 1024, Tk, 1024, 128, 1);

  // 4) query path: relu(conv3) -> relu(conv1) -> conv1
  wmma_conv_gemm<true, NP_Q><<<dim3(13, 3, B), NTHREADS, 0, stream>>>(
      qW1h, queriesP, qb1, q1P, 160, 96, Tq, 96, 192, 3);
  wmma_conv_gemm<true, NP_Q><<<dim3(13, 2, B), NTHREADS, 0, stream>>>(
      qW2h, q1P, qb2, q2bP, 80, 160, Tq, 192, 128, 1);
  wmma_conv_gemm<false, NP_Q><<<dim3(13, 2, B), NTHREADS, 0, stream>>>(
      qW3h, q2bP, qb3, qencP, 80, 96, Tq, 128, 128, 1);

  // 5) norms
  sumsq_pad<<<dim3(4, B), 256, 0, stream>>>(qencP, q2s, 80, Tq, NP_Q, 128, RAWM);
  sumsq_pad<<<dim3(1, B), 256, 0, stream>>>(kencP, k2s, 80, Tk, NP_K, 128, RAWN);

  // 6) attention scores: -5e-4 * ||q - k||^2 via WMMA QK
  wmma_qk_dist<NP_Q, NP_K><<<dim3(4, 13, B), NTHREADS, 0, stream>>>(
      qencP, kencP, q2s, k2s, rawP, 96, 128, 128);

  // 7) fused log_softmax + prior + mask + softmax
  softmax_rows<<<dim3(Tq, B), 256, 0, stream>>>(
      rawP, prior, mask, attn_out, logp_out, Tq, Tk);
}
